// RLBasedMLP_86105504350776
// MI455X (gfx1250) — matrix operand order repaired and checked by reading, not compiled, not run on hardware
//
#include <hip/hip_runtime.h>

typedef unsigned short ushort_t;
typedef __attribute__((ext_vector_type(16))) __bf16 v16bf;
typedef __attribute__((ext_vector_type(8)))  float  v8f;

constexpr int B_ = 8192;   // batch
constexpr int K_ = 4096;   // categories
constexpr int M_ = 64;     // samples
constexpr int H_ = 1024;   // hidden

__device__ __forceinline__ ushort_t f32_to_bf16_bits(float f) {
    union { float f; unsigned u; } x; x.f = f;
    unsigned r = x.u + 0x7FFFu + ((x.u >> 16) & 1u);   // round-to-nearest-even
    return (ushort_t)(r >> 16);
}
__device__ __forceinline__ float bf16_bits_to_f32(ushort_t h) {
    union { unsigned u; float f; } x; x.u = ((unsigned)h) << 16; return x.f;
}

// ---------------------------------------------------------------------------
// pack h = concat(h_real, h_imag) as bf16 [B, 2K]
// ---------------------------------------------------------------------------
__global__ __launch_bounds__(256) void pack_h_kernel(const float* __restrict__ hr,
                                                     const float* __restrict__ hi,
                                                     ushort_t* __restrict__ dst) {
    const size_t n = (size_t)B_ * 2 * K_;
    for (size_t i = (size_t)blockIdx.x * blockDim.x + threadIdx.x; i < n;
         i += (size_t)gridDim.x * blockDim.x) {
        size_t row = i >> 13;          // 2K = 8192 = 2^13
        size_t col = i & 8191;
        float v = (col < (size_t)K_) ? hr[row * K_ + col] : hi[row * K_ + (col - K_)];
        dst[i] = f32_to_bf16_bits(v);
    }
}

// f32 [Kd][N] row-major  ->  bf16 [N][Kd] (transposed, coalesced writes)
__global__ __launch_bounds__(256) void cvt_bf16_t_kernel(const float* __restrict__ W,
                                                         ushort_t* __restrict__ Wt,
                                                         int Kd, int N) {
    const size_t n = (size_t)N * Kd;
    for (size_t i = (size_t)blockIdx.x * blockDim.x + threadIdx.x; i < n;
         i += (size_t)gridDim.x * blockDim.x) {
        const size_t col = i / (size_t)Kd;   // n
        const size_t k   = i - col * Kd;     // k
        Wt[i] = f32_to_bf16_bits(W[k * N + col]);
    }
}

// ---------------------------------------------------------------------------
// WMMA bf16 GEMM:  out[M,N] = act(A[M,Kd] @ W[Kd,N] + bias), W given transposed.
// 256 threads = 8 waves (2x4 wave grid); block tile 128(M) x 128(N); K step 32.
// Each wave: 64 rows x 32 cols = 8 x v_wmma_f32_16x16x32_bf16 per K step.
// Double-buffered LDS, staged via GLOBAL_LOAD_ASYNC_TO_LDS_B128; since
// ASYNCcnt retires in order and each thread issues exactly 4 asyncs per tile,
// "s_wait_asynccnt 4" after issuing tile i+1 guarantees tile i has landed
// while tile i+1 is still in flight (DMA/WMMA overlap).
// ---------------------------------------------------------------------------
template <bool RELU, bool OUTBF16>
__global__ __launch_bounds__(256) void gemm_bf16_kernel(const ushort_t* __restrict__ A,
                                                        const ushort_t* __restrict__ Wt,
                                                        const float* __restrict__ bias,
                                                        void* __restrict__ outp,
                                                        int Kd, int N) {
    __shared__ ushort_t ldsA[2][128 * 32];   // [buf][row][k]
    __shared__ ushort_t ldsB[2][128 * 32];   // [buf][col][k]

    const int tid   = threadIdx.x;
    const int lane  = tid & 31;
    const int wave  = tid >> 5;
    const int waveM = wave >> 2;          // 0..1  (64-row strip)
    const int waveN = wave & 3;           // 0..3  (32-col strip)
    const int half  = lane >> 4;          // 0: lanes 0-15, 1: lanes 16-31
    const int l15   = lane & 15;
    const int rowBase = blockIdx.y * 128;
    const int colBase = blockIdx.x * 128;

    // staging assignment: thread t copies 32B of one A row and one B col
    const int sr  = tid >> 1;             // 0..127
    const int sko = (tid & 1) * 16;       // k offset 0 / 16
    const unsigned dstA[2] = {(unsigned)(size_t)&ldsA[0][sr * 32 + sko],
                              (unsigned)(size_t)&ldsA[1][sr * 32 + sko]};
    const unsigned dstB[2] = {(unsigned)(size_t)&ldsB[0][sr * 32 + sko],
                              (unsigned)(size_t)&ldsB[1][sr * 32 + sko]};
    const ushort_t* rowA = A  + (size_t)(rowBase + sr) * Kd + sko;
    const ushort_t* rowB = Wt + (size_t)(colBase + sr) * Kd + sko;

    // issue one tile's worth of async DMA (4x b128 per thread) into buffer b
    auto issue_tile = [&](int b, int k0) {
        const ushort_t* srcA = rowA + k0;
        const ushort_t* srcB = rowB + k0;
        asm volatile(
            "global_load_async_to_lds_b128 %0, %2, off\n\t"
            "global_load_async_to_lds_b128 %0, %2, off offset:16\n\t"
            "global_load_async_to_lds_b128 %1, %3, off\n\t"
            "global_load_async_to_lds_b128 %1, %3, off offset:16\n\t"
            :
            : "v"(dstA[b]), "v"(dstB[b]), "v"(srcA), "v"(srcB)
            : "memory");
    };

    union U16 { unsigned u[8]; v16bf v; };
    v8f acc[4][2] = {};

    issue_tile(0, 0);                      // prologue: prime buffer 0
    int buf = 0;
    for (int k0 = 0; k0 < Kd; k0 += 32) {
        if (k0 + 32 < Kd) {
            issue_tile(buf ^ 1, k0 + 32);  // prefetch next tile into other buffer
            __builtin_prefetch(rowA + k0 + 64, 0, 1);   // global_prefetch_b8 (L2 warm)
            asm volatile("s_wait_asynccnt 4" ::: "memory");   // tile k0 landed
        } else {
            asm volatile("s_wait_asynccnt 0" ::: "memory");
        }
        __syncthreads();

        // ---- 2 B fragments (kept live), 4 streamed A fragments, 8 WMMA ----
        U16 bfr[2];
#pragma unroll
        for (int ni = 0; ni < 2; ++ni) {
            const int cbase = (waveN * 32 + ni * 16 + l15) * 32 + half * 16;
#pragma unroll
            for (int p = 0; p < 8; ++p)
                bfr[ni].u[p] = *(const unsigned*)&ldsB[buf][cbase + 2 * p];
        }
#pragma unroll
        for (int mi = 0; mi < 4; ++mi) {
            U16 af;
            const int arow = (waveM * 64 + mi * 16 + l15) * 32;
#pragma unroll
            for (int p = 0; p < 8; ++p) {
                const int k = half * 16 + 2 * p;
                af.u[p] = *(const unsigned*)&ldsA[buf][arow + k];
            }
#pragma unroll
            for (int ni = 0; ni < 2; ++ni)
                acc[mi][ni] = __builtin_amdgcn_wmma_f32_16x16x32_bf16(
                    false, af.v, false, bfr[ni].v, (short)0, acc[mi][ni], false, false);
        }
        __syncthreads();   // all waves done reading buf before it is overwritten
        buf ^= 1;
    }

    // ---- epilogue: bias (+ReLU), store f32 or bf16 ----
#pragma unroll
    for (int mi = 0; mi < 4; ++mi) {
#pragma unroll
        for (int ni = 0; ni < 2; ++ni) {
            const int col = colBase + waveN * 32 + ni * 16 + l15;
            const float bv = bias[col];
#pragma unroll
            for (int e = 0; e < 8; ++e) {
                const int row = rowBase + waveM * 64 + mi * 16 + half * 8 + e;
                float v = acc[mi][ni][e] + bv;
                if (RELU) v = v > 0.f ? v : 0.f;
                if (OUTBF16)
                    ((ushort_t*)outp)[(size_t)row * N + col] = f32_to_bf16_bits(v);
                else
                    ((float*)outp)[(size_t)row * N + col] = v;
            }
        }
    }
}

// ---------------------------------------------------------------------------
// value head layer 2: value[row] = sum_k vh_bf16[row,k] * Wv2[k] + b_v2
// ---------------------------------------------------------------------------
__global__ __launch_bounds__(128) void value_kernel(const ushort_t* __restrict__ vh,
                                                    const float* __restrict__ Wv2,
                                                    const float* __restrict__ bv2,
                                                    float* __restrict__ val) {
    __shared__ float red[128];
    const int row = blockIdx.x, tid = threadIdx.x;
    float s = 0.f;
    for (int k = tid; k < 512; k += 128)
        s += bf16_bits_to_f32(vh[(size_t)row * 512 + k]) * Wv2[k];
    red[tid] = s;
    __syncthreads();
    for (int st = 64; st > 0; st >>= 1) {
        if (tid < st) red[tid] += red[tid + st];
        __syncthreads();
    }
    if (tid == 0) val[row] = red[0] + bv2[0];
}

// ---------------------------------------------------------------------------
// Gumbel-top-64 + ordered log-prob + gather of h_sel (bf16) — one block/row
// ---------------------------------------------------------------------------
__global__ __launch_bounds__(256) void sample_kernel(const float* __restrict__ logits,
                                                     const float* __restrict__ u,
                                                     const float* __restrict__ h_real,
                                                     const float* __restrict__ h_imag,
                                                     int* __restrict__ idx_out,
                                                     float* __restrict__ logp_out,
                                                     ushort_t* __restrict__ h_sel) {
    __shared__ float keys[K_];
    __shared__ float sKey[256];
    __shared__ int   sIdx[256];
    __shared__ int   chosen[M_];
    __shared__ float sc[4];    // m, sumexp, cums, logp

    const int row = blockIdx.x, tid = threadIdx.x;
    const float* lrow = logits + (size_t)row * K_;
    const float* urow = u + (size_t)row * K_;

    // pass 1: row max of logits
    float lmax = -3.4e38f;
    for (int k = tid; k < K_; k += 256) lmax = fmaxf(lmax, lrow[k]);
    sKey[tid] = lmax;
    __syncthreads();
    for (int s = 128; s > 0; s >>= 1) {
        if (tid < s) sKey[tid] = fmaxf(sKey[tid], sKey[tid + s]);
        __syncthreads();
    }
    if (tid == 0) { sc[0] = sKey[0]; sc[2] = 0.f; sc[3] = 0.f; }
    __syncthreads();
    const float m = sc[0];

    // pass 2: Gumbel keys into LDS + sumexp(logits - m)
    float lsum = 0.f;
    for (int k = tid; k < K_; k += 256) {
        const float lg = lrow[k];
        const float uc = fminf(fmaxf(urow[k], 1e-9f), 1.f - 1e-9f);
        keys[k] = lg - __logf(-__logf(uc));
        lsum += __expf(lg - m);
    }
    sKey[tid] = lsum;
    __syncthreads();
    for (int s = 128; s > 0; s >>= 1) {
        if (tid < s) sKey[tid] += sKey[tid + s];
        __syncthreads();
    }
    if (tid == 0) sc[1] = sKey[0];
    __syncthreads();
    const float sumexp = sc[1];

    // 64 sequential argmax rounds (ordered sample)
    for (int t = 0; t < M_; ++t) {
        float bk = -3.4e38f;
        int bi = 0;
        for (int k = tid; k < K_; k += 256) {
            const float kk = keys[k];
            if (kk > bk) { bk = kk; bi = k; }
        }
        sKey[tid] = bk; sIdx[tid] = bi;
        __syncthreads();
        for (int s = 128; s > 0; s >>= 1) {
            if (tid < s && sKey[tid + s] > sKey[tid]) {
                sKey[tid] = sKey[tid + s];
                sIdx[tid] = sIdx[tid + s];
            }
            __syncthreads();
        }
        if (tid == 0) {
            const int b = sIdx[0];
            chosen[t] = b;
            const float sel   = lrow[b];
            const float denom = m + __logf(fmaxf(sumexp - sc[2], 1e-30f));
            sc[3] += sel - denom;
            sc[2] += __expf(sel - m);
            keys[b] = -3.4e38f;
        }
        __syncthreads();
    }

    if (tid < M_) {
        const int b = chosen[tid];
        idx_out[(size_t)row * M_ + tid] = b;
        h_sel[(size_t)row * 2 * M_ + tid]      = f32_to_bf16_bits(h_real[(size_t)row * K_ + b]);
        h_sel[(size_t)row * 2 * M_ + M_ + tid] = f32_to_bf16_bits(h_imag[(size_t)row * K_ + b]);
    }
    if (tid == 0) logp_out[row] = sc[3];
}

// ---------------------------------------------------------------------------
extern "C" void kernel_launch(void* const* d_in, const int* in_sizes, int n_in,
                              void* d_out, int out_size, void* d_ws, size_t ws_size,
                              hipStream_t stream) {
    const float* h_real = (const float*)d_in[0];
    const float* h_imag = (const float*)d_in[1];
    const float* u      = (const float*)d_in[2];
    const float* W_p1 = (const float*)d_in[3];  const float* b_p1 = (const float*)d_in[4];
    const float* W_p2 = (const float*)d_in[5];  const float* b_p2 = (const float*)d_in[6];
    const float* W_v1 = (const float*)d_in[7];  const float* b_v1 = (const float*)d_in[8];
    const float* W_v2 = (const float*)d_in[9];  const float* b_v2 = (const float*)d_in[10];
    const float* W_n1 = (const float*)d_in[11]; const float* b_n1 = (const float*)d_in[12];
    const float* W_n2 = (const float*)d_in[13]; const float* b_n2 = (const float*)d_in[14];

    // output regions (tuple concatenated flat): output[B*K] f32, idx[B*M] i32,
    // log_probs[B] f32, value[B] f32
    float* outF      = (float*)d_out;
    float* out_main  = outF;
    int*   idx_out   = (int*)(outF + (size_t)B_ * K_);
    float* logp_out  = outF + (size_t)B_ * K_ + (size_t)B_ * M_;
    float* value_out = logp_out + B_;

    // workspace carve-up (256B aligned chunks)
    char* ws = (char*)d_ws;
    size_t off = 0;
    auto carve = [&](size_t bytes) -> void* {
        void* p = ws + off;
        off += (bytes + 255) & ~(size_t)255;
        return p;
    };
    ushort_t* h_bf   = (ushort_t*)carve((size_t)B_ * 2 * K_ * 2);      // 128 MB
    ushort_t* Wp1t   = (ushort_t*)carve((size_t)2 * K_ * H_ * 2);      // 16 MB  [H][2K]
    ushort_t* Wp2t   = (ushort_t*)carve((size_t)H_ * K_ * 2);          // 8 MB   [K][H]
    ushort_t* Wv1t   = (ushort_t*)carve((size_t)2 * K_ * (H_ / 2) * 2);// 8 MB   [H/2][2K]
    ushort_t* Wn1t   = (ushort_t*)carve((size_t)2 * M_ * H_ * 2);      // 256 KB [H][2M]
    ushort_t* Wn2t   = (ushort_t*)carve((size_t)H_ * K_ * 2);          // 8 MB   [K][H]
    ushort_t* hidden = (ushort_t*)carve((size_t)B_ * H_ * 2);          // 16 MB
    ushort_t* vhid   = (ushort_t*)carve((size_t)B_ * (H_ / 2) * 2);    // 8 MB
    float*    logits = (float*)   carve((size_t)B_ * K_ * 4);          // 128 MB
    ushort_t* h_sel  = (ushort_t*)carve((size_t)B_ * 2 * M_ * 2);      // 2 MB
    ushort_t* nhid   = (ushort_t*)carve((size_t)B_ * H_ * 2);          // 16 MB

    // ---- stage 0: precision packing + weight transposition ----
    pack_h_kernel<<<4096, 256, 0, stream>>>(h_real, h_imag, h_bf);
    cvt_bf16_t_kernel<<<2048, 256, 0, stream>>>(W_p1, Wp1t, 2 * K_, H_);
    cvt_bf16_t_kernel<<<2048, 256, 0, stream>>>(W_p2, Wp2t, H_, K_);
    cvt_bf16_t_kernel<<<2048, 256, 0, stream>>>(W_v1, Wv1t, 2 * K_, H_ / 2);
    cvt_bf16_t_kernel<<<256, 256, 0, stream>>>(W_n1, Wn1t, 2 * M_, H_);
    cvt_bf16_t_kernel<<<2048, 256, 0, stream>>>(W_n2, Wn2t, H_, K_);

    // ---- policy + value MLPs (WMMA, double-buffered async-LDS staging) ----
    gemm_bf16_kernel<true, true><<<dim3(H_ / 128, B_ / 128), 256, 0, stream>>>(
        h_bf, Wp1t, b_p1, hidden, 2 * K_, H_);                       // relu(h@Wp1+b)
    gemm_bf16_kernel<true, true><<<dim3((H_ / 2) / 128, B_ / 128), 256, 0, stream>>>(
        h_bf, Wv1t, b_v1, vhid, 2 * K_, H_ / 2);                     // relu(h@Wv1+b)
    gemm_bf16_kernel<false, false><<<dim3(K_ / 128, B_ / 128), 256, 0, stream>>>(
        hidden, Wp2t, b_p2, logits, H_, K_);                         // logits (f32)
    value_kernel<<<B_, 128, 0, stream>>>(vhid, W_v2, b_v2, value_out);

    // ---- Gumbel top-M sampling, log-probs, gather ----
    sample_kernel<<<B_, 256, 0, stream>>>(logits, u, h_real, h_imag,
                                          idx_out, logp_out, h_sel);

    // ---- decoder MLP on selected features (WMMA) ----
    gemm_bf16_kernel<true, true><<<dim3(H_ / 128, B_ / 128), 256, 0, stream>>>(
        h_sel, Wn1t, b_n1, nhid, 2 * M_, H_);                        // relu(h_sel@Wn1+b)
    gemm_bf16_kernel<false, false><<<dim3(K_ / 128, B_ / 128), 256, 0, stream>>>(
        nhid, Wn2t, b_n2, out_main, H_, K_);                         // final output (f32)
}
